// CLVPAttention_42666205119167
// MI455X (gfx1250) — compile-verified, matrix operand order checked by reading, not run on hardware
//
#include <hip/hip_runtime.h>
#include <math.h>

typedef __bf16 bf16;
typedef __attribute__((ext_vector_type(8)))  __bf16 v8bf;
typedef __attribute__((ext_vector_type(16))) __bf16 v16bf;
typedef __attribute__((ext_vector_type(8)))  float  v8f;
typedef __attribute__((ext_vector_type(4)))  int    v4i;

// Problem constants (from reference)
constexpr int Bb  = 2;
constexpr int T   = 2048;
constexpr int E   = 768;
constexpr int H   = 12;
constexpr int HD  = 64;          // E / H
constexpr int M   = Bb * T;      // 4096 rows of activations
constexpr float SCALE_Q = 0.125f; // HD^-0.5

// Async global->LDS path (gfx1250, ASYNCcnt-tracked). Guarded so the file
// still compiles (synchronous staging fallback) if this toolchain does not
// declare the builtins.
#if __has_builtin(__builtin_amdgcn_global_load_async_to_lds_b128) && \
    __has_builtin(__builtin_amdgcn_s_wait_asynccnt)
#define USE_ASYNC_LDS 1
#else
#define USE_ASYNC_LDS 0
#endif

// ---------------------------------------------------------------------------
// helpers
// ---------------------------------------------------------------------------
__device__ inline v8f v8f_zero() {
    v8f z;
#pragma unroll
    for (int i = 0; i < 8; ++i) z[i] = 0.f;
    return z;
}

// A/B fragment for v_wmma_f32_16x16x32_bf16.
// Per ISA: lane L holds row (L&15); K elements e<8 -> k = half*8+e,
// e>=8 -> k = 16 + half*8 + (e-8)  (half = L>>4). Both halves are two
// contiguous 8x bf16 (16B) chunks when K is contiguous in memory.
__device__ inline v16bf load_frag(const bf16* rowk0, int half) {
    union { v16bf v; v8bf h[2]; } u;
    u.h[0] = *(const v8bf*)(rowk0 + half * 8);
    u.h[1] = *(const v8bf*)(rowk0 + 16 + half * 8);
    return u.v;
}

__device__ inline v8f wmma_bf16(v16bf a, v16bf b, v8f c) {
    return __builtin_amdgcn_wmma_f32_16x16x32_bf16(
        /*neg_a=*/false, a, /*neg_b=*/false, b,
        /*c_mod=*/(short)0, c, /*reuse_a=*/false, /*reuse_b=*/false);
}

// Stage one 16B chunk of a shared tile into LDS.
__device__ inline void stage16(const bf16* g, bf16* l) {
#if USE_ASYNC_LDS
    // param types per hipcc: (int4 __device__*, int4 lds*, imm offset, imm cpol);
    // pass generic int4* and let clang insert the addrspacecasts.
    __builtin_amdgcn_global_load_async_to_lds_b128((v4i*)g, (v4i*)l,
                                                   /*offset=*/0, /*cpol=*/0);
#else
    *(v8bf*)l = *(const v8bf*)g;
#endif
}

// ---------------------------------------------------------------------------
// f32 -> bf16 conversion (grid-stride)
// ---------------------------------------------------------------------------
__global__ void to_bf16_kernel(const float* __restrict__ in, bf16* __restrict__ out, int n) {
    int i = blockIdx.x * blockDim.x + threadIdx.x;
    for (; i < n; i += gridDim.x * blockDim.x) out[i] = (bf16)in[i];
}

// ---------------------------------------------------------------------------
// WMMA GEMM:  C[m][n] = scale * sum_k A[m][k] * W[n][k]   (A: MxE, W: ExE row-major)
// Block = 8 waves x (32M x 64N) tiles over the SAME 64 N-columns; the shared
// 64x32 W tile is double-buffer async-DMA'd into LDS (8x reuse), A-fragments
// stream from global. Each B-fragment feeds 2 WMMAs.
// MODE 0: RoPE epilogue -> bf16 [bh][t][HD]           (Q / K)
// MODE 1: RoPE epilogue -> bf16 [bh][d][t] transposed (V)
// MODE 2: f32 + bias    -> row-major [m][n]           (output projection)
// ---------------------------------------------------------------------------
template <int MODE>
__global__ __launch_bounds__(256) void gemm_wmma_kernel(
    const bf16* __restrict__ A, const bf16* __restrict__ W,
    bf16* __restrict__ outb, float* __restrict__ outf,
    const float* __restrict__ freqs, const float* __restrict__ bias, float scale)
{
    __shared__ bf16 Wt[2][64 * 32];   // double-buffered 64(N) x 32(K) tile, 8KB

    const int tid  = threadIdx.x;
    const int wave = tid >> 5;
    const int lane = tid & 31;
    const int l16  = lane & 15;
    const int half = lane >> 4;
    const int m_base = (blockIdx.y * 8 + wave) * 32;
    const int n_base = blockIdx.x * 64;      // == head * HD for MODE 0/1

    // staging: each of 256 threads copies one 16B chunk per 64x32 tile (4KB)
    const int crow = tid >> 2;               // 0..63 : N row within tile
    const int cchk = (tid & 3) * 8;          // bf16 offset of 16B chunk in K
    const bf16* gsrc = W + (size_t)(n_base + crow) * E + cchk;

    v8f acc[2][4];
#pragma unroll
    for (int mi = 0; mi < 2; ++mi)
#pragma unroll
        for (int t = 0; t < 4; ++t) acc[mi][t] = v8f_zero();

    const bf16* arow0 = A + (size_t)(m_base + l16) * E;
    const bf16* arow1 = A + (size_t)(m_base + 16 + l16) * E;

    stage16(gsrc, &Wt[0][crow * 32 + cchk]);  // prologue: tile 0 -> buf 0

    for (int k0 = 0; k0 < E; k0 += 32) {
        const int cur = (k0 >> 5) & 1;
        if (k0 + 32 < E)
            stage16(gsrc + k0 + 32, &Wt[cur ^ 1][crow * 32 + cchk]);
#if USE_ASYNC_LDS
        if (k0 + 32 < E) __builtin_amdgcn_s_wait_asynccnt(1);  // current tile done
        else             __builtin_amdgcn_s_wait_asynccnt(0);
#endif
        __syncthreads();                      // all waves' chunks visible

        if (k0 + 64 < E) {
            __builtin_prefetch(arow0 + k0 + 64, 0, 1);  // global_prefetch_b8
            __builtin_prefetch(arow1 + k0 + 64, 0, 1);
        }
        v16bf a0 = load_frag(arow0 + k0, half);
        v16bf a1 = load_frag(arow1 + k0, half);
#pragma unroll
        for (int t = 0; t < 4; ++t) {
            v16bf b = load_frag(&Wt[cur][(t * 16 + l16) * 32], half);
            acc[0][t] = wmma_bf16(a0, b, acc[0][t]);
            acc[1][t] = wmma_bf16(a1, b, acc[1][t]);
        }
        __syncthreads();                      // done reading before overwrite
    }

    if (MODE == 2) {
#pragma unroll
        for (int mi = 0; mi < 2; ++mi)
#pragma unroll
            for (int t = 0; t < 4; ++t)
#pragma unroll
                for (int j = 0; j < 8; ++j) {
                    int row = m_base + mi * 16 + j + 8 * half;
                    int col = n_base + t * 16 + l16;
                    float v = acc[mi][t][j] * scale + (bias ? bias[col] : 0.f);
                    outf[(size_t)row * E + col] = v;
                }
        return;
    }

    // MODE 0 / 1: fused rotary. This wave's 64 N-columns are one whole head,
    // so the d <-> d+-16 mixing stays inside acc[mi][0]/acc[mi][1], same lane.
    const int h = blockIdx.x;  // head index
#pragma unroll
    for (int mi = 0; mi < 2; ++mi)
#pragma unroll
        for (int j = 0; j < 8; ++j) {
            int row = m_base + mi * 16 + j + 8 * half;  // token row in [0, M)
            int b   = row / T;
            int tok = row - b * T;
            int bh  = b * H + h;
            const float* f = freqs + (size_t)tok * 32;

            float v0 = acc[mi][0][j] * scale;  // d = l16        (< 16)
            float v1 = acc[mi][1][j] * scale;  // d = 16 + l16   (< 32)
            float v2 = acc[mi][2][j] * scale;  // untouched by RoPE
            float v3 = acc[mi][3][j] * scale;

            float c0 = __cosf(f[l16]),      s0 = __sinf(f[l16]);
            float c1 = __cosf(f[16 + l16]), s1 = __sinf(f[16 + l16]);
            float r0 = v0 * c0 - v1 * s0;      // -x2 branch
            float r1 = v1 * c1 + v0 * s1;      // +x1 branch

            if (MODE == 0) {
                bf16* dst = outb + ((size_t)bh * T + tok) * HD;
                dst[l16]      = (bf16)r0;
                dst[16 + l16] = (bf16)r1;
                dst[32 + l16] = (bf16)v2;
                dst[48 + l16] = (bf16)v3;
            } else {
                // transposed: vt[(bh*HD + d)*T + tok]
                size_t base = (size_t)bh * HD * T + tok;
                outb[base + (size_t)(l16)      * T] = (bf16)r0;
                outb[base + (size_t)(16 + l16) * T] = (bf16)r1;
                outb[base + (size_t)(32 + l16) * T] = (bf16)v2;
                outb[base + (size_t)(48 + l16) * T] = (bf16)v3;
            }
        }
}

// ---------------------------------------------------------------------------
// Flash attention: one wave per 16-token q block per head-batch.
// S = Q*K^T (16x32 per step, 4 wmma), online softmax via LDS + shfl,
// O += P*V (4 wmma into 16x64 f32 accumulator).
// ---------------------------------------------------------------------------
__global__ __launch_bounds__(256) void attn_wmma_kernel(
    const bf16* __restrict__ qh, const bf16* __restrict__ kh,
    const bf16* __restrict__ vt, bf16* __restrict__ ctx)
{
    __shared__ float Sls[8][16 * 32];
    __shared__ bf16  Pls[8][16 * 32];

    const int wave = threadIdx.x >> 5;
    const int lane = threadIdx.x & 31;
    const int l16  = lane & 15;
    const int half = lane >> 4;

    const int wg = blockIdx.x * 8 + wave;
    const int qb = (wg % (T / 16)) * 16;
    const int bh = wg / (T / 16);

    float* S = Sls[wave];
    bf16*  P = Pls[wave];

    const bf16* qrow = qh + ((size_t)bh * T + qb + l16) * HD;
    v16bf aq0 = load_frag(qrow, half);
    v16bf aq1 = load_frag(qrow + 32, half);

    const bf16* kbase = kh + (size_t)bh * T * HD;
    const bf16* vbase = vt + (size_t)bh * HD * T;

    float m_i = -1e30f, l_i = 0.f;
    v8f o[4];
#pragma unroll
    for (int t = 0; t < 4; ++t) o[t] = v8f_zero();

    for (int kb = 0; kb < T; kb += 32) {
        // ---- scores: 16 q rows x 32 k rows ----
#pragma unroll
        for (int st = 0; st < 2; ++st) {
            const bf16* krow = kbase + (size_t)(kb + st * 16 + l16) * HD;
            v8f s = v8f_zero();
            s = wmma_bf16(aq0, load_frag(krow, half), s);
            s = wmma_bf16(aq1, load_frag(krow + 32, half), s);
#pragma unroll
            for (int j = 0; j < 8; ++j)
                S[(j + 8 * half) * 32 + st * 16 + l16] = s[j];
        }
        asm volatile("s_wait_dscnt 0" ::: "memory");  // wave-local LDS RAW

        // ---- online softmax: lane handles row l16, cols [half*16, half*16+16) ----
        const float* srow = S + l16 * 32 + half * 16;
        float tm = srow[0];
#pragma unroll
        for (int i = 1; i < 16; ++i) tm = fmaxf(tm, srow[i]);
        tm = fmaxf(tm, __shfl_xor(tm, 16, 32));
        float m_new = fmaxf(m_i, tm);
        float corr  = __expf(m_i - m_new);
        float lsum  = 0.f;
#pragma unroll
        for (int i = 0; i < 16; ++i) {
            float p = __expf(srow[i] - m_new);
            lsum += p;
            P[l16 * 32 + half * 16 + i] = (bf16)p;
        }
        lsum += __shfl_xor(lsum, 16, 32);
        l_i = l_i * corr + lsum;
        m_i = m_new;
        asm volatile("s_wait_dscnt 0" ::: "memory");

        // ---- rescale accumulator rows, then O += P * V ----
#pragma unroll
        for (int j = 0; j < 8; ++j) {
            float cj = __shfl(corr, j + 8 * half, 32);
#pragma unroll
            for (int t = 0; t < 4; ++t) o[t][j] *= cj;
        }
        v16bf ap = load_frag(P + l16 * 32, half);
#pragma unroll
        for (int t = 0; t < 4; ++t) {
            v16bf bv = load_frag(vbase + (size_t)(t * 16 + l16) * T + kb, half);
            o[t] = wmma_bf16(ap, bv, o[t]);
        }
    }

    // ---- normalize and write context in [b][t][h*HD + d] bf16 ----
    float inv = 1.f / l_i;
    const int b = bh / H, h = bh - b * H;
#pragma unroll
    for (int j = 0; j < 8; ++j) {
        float sj = __shfl(inv, j + 8 * half, 32);
        int tok  = qb + j + 8 * half;
        size_t base = ((size_t)(b * T + tok)) * E + (size_t)h * HD;
#pragma unroll
        for (int t = 0; t < 4; ++t)
            ctx[base + t * 16 + l16] = (bf16)(o[t][j] * sj);
    }
}

// ---------------------------------------------------------------------------
// launch
// ---------------------------------------------------------------------------
extern "C" void kernel_launch(void* const* d_in, const int* in_sizes, int n_in,
                              void* d_out, int out_size, void* d_ws, size_t ws_size,
                              hipStream_t stream) {
    (void)in_sizes; (void)n_in; (void)out_size; (void)ws_size;

    const float* hs   = (const float*)d_in[0];
    const float* rope = (const float*)d_in[1];
    const float* Wq   = (const float*)d_in[2];
    const float* Wk   = (const float*)d_in[3];
    const float* Wv   = (const float*)d_in[4];
    const float* Wo   = (const float*)d_in[5];
    const float* bo   = (const float*)d_in[6];

    // workspace carve (all bf16), 256B aligned
    size_t off = 0;
    auto carve = [&](size_t bytes) {
        void* p = (char*)d_ws + off;
        off += (bytes + 255) & ~(size_t)255;
        return p;
    };
    bf16* Xbf  = (bf16*)carve((size_t)M * E * 2);
    bf16* Wqb  = (bf16*)carve((size_t)E * E * 2);
    bf16* Wkb  = (bf16*)carve((size_t)E * E * 2);
    bf16* Wvb  = (bf16*)carve((size_t)E * E * 2);
    bf16* Wob  = (bf16*)carve((size_t)E * E * 2);
    bf16* qhb  = (bf16*)carve((size_t)Bb * H * T * HD * 2);
    bf16* khb  = (bf16*)carve((size_t)Bb * H * T * HD * 2);
    bf16* vtb  = (bf16*)carve((size_t)Bb * H * T * HD * 2);
    bf16* ctxb = (bf16*)carve((size_t)M * E * 2);

    to_bf16_kernel<<<512, 256, 0, stream>>>(hs, Xbf, M * E);
    to_bf16_kernel<<<512, 256, 0, stream>>>(Wq, Wqb, E * E);
    to_bf16_kernel<<<512, 256, 0, stream>>>(Wk, Wkb, E * E);
    to_bf16_kernel<<<512, 256, 0, stream>>>(Wv, Wvb, E * E);
    to_bf16_kernel<<<512, 256, 0, stream>>>(Wo, Wob, E * E);

    dim3 gproj(E / 64, M / 256);   // (12, 16): 8 waves of 32x64 tiles per block
    gemm_wmma_kernel<0><<<gproj, 256, 0, stream>>>(Xbf, Wqb, qhb, nullptr, rope, nullptr, SCALE_Q);
    gemm_wmma_kernel<0><<<gproj, 256, 0, stream>>>(Xbf, Wkb, khb, nullptr, rope, nullptr, 1.f);
    gemm_wmma_kernel<1><<<gproj, 256, 0, stream>>>(Xbf, Wvb, vtb, nullptr, rope, nullptr, 1.f);

    attn_wmma_kernel<<<(Bb * H * (T / 16)) / 8, 256, 0, stream>>>(qhb, khb, vtb, ctxb);

    gemm_wmma_kernel<2><<<gproj, 256, 0, stream>>>(ctxb, Wob, nullptr, (float*)d_out, nullptr, bo, 1.f);
}
